// RandomizedAttentionPolicy_23553600651913
// MI455X (gfx1250) — matrix operand order repaired and checked by reading, not compiled, not run on hardware
//
#include <hip/hip_runtime.h>
#include <math.h>

typedef __attribute__((ext_vector_type(16))) _Float16 v16h;
typedef __attribute__((ext_vector_type(8)))  _Float16 v8h;
typedef __attribute__((ext_vector_type(8)))  float    v8f;

#define ACT_NONE 0
#define ACT_LRELU 1
#define ACT_TANH 2

// store modes for the fused GEMM epilogue
#define ST_F16_ROW 0
#define ST_F16_TR  1
#define ST_F32_ROW 2

namespace {

struct Params {
  const float* x;        // [B,64,64]
  const float* rnd;      // [B,64]
  const int*   idx;      // [B,64]
  const float* embed_w;  const float* embed_b;
  const float* ia_wq; const float* ia_bq; const float* ia_wk; const float* ia_bk;
  const float* ia_wv; const float* ia_bv; const float* ia_wo; const float* ia_bo;
  const float* c1_wq; const float* c1_bq; const float* c1_wk; const float* c1_bk;
  const float* c1_wv; const float* c1_bv; const float* c1_wo; const float* c1_bo;
  const float* c2_wq; const float* c2_bq; const float* c2_wk; const float* c2_bk;
  const float* c2_wv; const float* c2_bv; const float* c2_wo; const float* c2_bo;
  const float* msg_w; const float* msg_b; const float* st_w; const float* st_b;
  const float* mean_w; const float* mean_b; const float* ls_w; const float* ls_b;
  float* out;            // mean flat [B*64*8], then std flat [B*64*8]
};

// ---- WMMA fragment loaders (wave32, v_wmma_f32_16x16x32_f16) --------------
// A-frag (16x32): lane holds row r0+lane%16; ISA K striping:
//   k = kb + (lane/16)*8 + h + (h&8)   (two 16B contiguous runs -> 2x b128)
__device__ __forceinline__ v16h fragA(const _Float16* s, int r0, int kb){
  const int l = threadIdx.x & 31;
  const int m = l & 15, g = l >> 4;
  const _Float16* p = s + (r0 + m)*64 + kb + g*8;
  v16h f;
#pragma unroll
  for (int h = 0; h < 16; ++h) f[h] = p[h + (h & 8)];
  return f;
}
// B-frag (32x16) from row-major [N][K] storage: lane holds column n0+lane%16;
// ISA B striping matches A: k = kb + (lane/16)*8 + h + (h&8)
// -> two 16B contiguous runs -> 2x ds_load_b128.
__device__ __forceinline__ v16h fragB(const _Float16* s, int n0, int kb){
  const int l = threadIdx.x & 31;
  const int n = l & 15, g = l >> 4;
  const _Float16* p = s + (n0 + n)*64 + kb + g*8;
  v16h f;
#pragma unroll
  for (int h = 0; h < 16; ++h) f[h] = p[h + (h & 8)];
  return f;
}

// Fused 64x64 GEMM: D = act(A[64x64] x B([N][K] row-major) + bias), stored per MODE.
//  - bias[n] is one load per lane per tile (acc lane = output column n)
//  - ST_F16_TR: lane's 8 rows are contiguous in the transposed dest -> ds_store_b128
template<int ACT, int MODE, bool HASBIAS>
__device__ __forceinline__ void mmF(const _Float16* A, const _Float16* B,
                                    const float* bias, void* dstv){
  const int wave = threadIdx.x >> 5;
  const int l    = threadIdx.x & 31;
#pragma unroll
  for (int t = 0; t < 2; ++t){
    const int tile = wave*2 + t;
    const int mt = (tile >> 2) << 4;
    const int nt = (tile &  3) << 4;
    v8f acc = {};
#pragma unroll
    for (int kb = 0; kb < 64; kb += 32){
      v16h a = fragA(A, mt, kb);
      v16h b = fragB(B, nt, kb);
      acc = __builtin_amdgcn_wmma_f32_16x16x32_f16(false, a, false, b,
                                                   (short)0, acc, false, false);
    }
    const int n     = nt + (l & 15);
    const int mbase = mt + ((l >> 4) << 3);
    const float bb  = HASBIAS ? bias[n] : 0.f;
    float r[8];
#pragma unroll
    for (int k = 0; k < 8; ++k){
      float v = acc[k] + bb;
      if (ACT == ACT_LRELU)      v = (v > 0.f) ? v : 0.2f * v;
      else if (ACT == ACT_TANH)  v = tanhf(v);
      r[k] = v;
    }
    if (MODE == ST_F32_ROW){
      float* D = (float*)dstv;
#pragma unroll
      for (int k = 0; k < 8; ++k) D[(mbase + k)*64 + n] = r[k];
    } else if (MODE == ST_F16_TR){
      _Float16* D = (_Float16*)dstv;
      v8h pk;
#pragma unroll
      for (int k = 0; k < 8; ++k) pk[k] = (_Float16)r[k];
      *(v8h*)(D + n*64 + mbase) = pk;           // 16B contiguous -> b128 store
    } else {
      _Float16* D = (_Float16*)dstv;
#pragma unroll
      for (int k = 0; k < 8; ++k) D[(mbase + k)*64 + n] = (_Float16)r[k];
    }
  }
}

// stage a 64x64 f32 global matrix into LDS as f16 (256 threads x 16 elems)
__device__ __forceinline__ void stageW(const float* g, _Float16* s){
  const int base = threadIdx.x * 16;
#pragma unroll
  for (int e = 0; e < 16; e += 4){
    const float4 v = *(const float4*)(g + base + e);
    s[base + e + 0] = (_Float16)v.x;
    s[base + e + 1] = (_Float16)v.y;
    s[base + e + 2] = (_Float16)v.z;
    s[base + e + 3] = (_Float16)v.w;
  }
}

// prefetch next weight matrix into cache (global_prefetch_b8)
__device__ __forceinline__ void pf(const float* g){
  __builtin_prefetch(g + threadIdx.x * 16, 0, 3);
}

__device__ __forceinline__ void softmaxPlain(const float* S, _Float16* W16){
  const int i = threadIdx.x;
  if (i < 64){
    const float sc = 0.125f; // 1/sqrt(64)
    const float4* Sr = (const float4*)(S + i*64);
    float mx = -1e30f;
    for (int j = 0; j < 16; ++j){
      float4 s4 = Sr[j];
      mx = fmaxf(mx, fmaxf(fmaxf(s4.x, s4.y), fmaxf(s4.z, s4.w)) * sc);
    }
    float sum = 0.f;
    for (int j = 0; j < 16; ++j){
      float4 s4 = Sr[j];
      sum += __expf(s4.x*sc - mx) + __expf(s4.y*sc - mx)
           + __expf(s4.z*sc - mx) + __expf(s4.w*sc - mx);
    }
    const float inv = 1.f / sum;
    for (int j = 0; j < 16; ++j){
      float4 s4 = Sr[j];
      W16[i*64 + j*4 + 0] = (_Float16)(__expf(s4.x*sc - mx) * inv);
      W16[i*64 + j*4 + 1] = (_Float16)(__expf(s4.y*sc - mx) * inv);
      W16[i*64 + j*4 + 2] = (_Float16)(__expf(s4.z*sc - mx) * inv);
      W16[i*64 + j*4 + 3] = (_Float16)(__expf(s4.w*sc - mx) * inv);
    }
  }
}

// masked consensus softmax: s = (m*G + qq.bk)*sc ; out = m * softmax(s)
__device__ __forceinline__ void softmaxMasked(const float* S, const _Float16* QQ,
                                              const float* bk, const float* rnd_s,
                                              const float* a_s, int inv_mask,
                                              _Float16* W16){
  const int i = threadIdx.x;
  if (i < 64){
    const float sc = 0.125f;
    float qb = 0.f;
    const v16h* Qr = (const v16h*)(QQ + i*64);   // 32B chunks -> ds_load_b128 pairs
#pragma unroll
    for (int c = 0; c < 4; ++c){
      v16h qv = Qr[c];
#pragma unroll
      for (int u = 0; u < 16; ++u) qb += (float)qv[u] * bk[c*16 + u];
    }
    const float ai = a_s[i];
    const int   gi = rnd_s[i] > ai;
    const float4* Sr = (const float4*)(S + i*64);
    float mx = -1e30f;
    for (int j4 = 0; j4 < 16; ++j4){
      float4 s4 = Sr[j4];
      const float* sv = (const float*)&s4;
#pragma unroll
      for (int u = 0; u < 4; ++u){
        float m = (gi || (rnd_s[j4*4+u] > ai)) ? 1.f : 0.f;
        if (inv_mask) m = 1.f - m;
        mx = fmaxf(mx, (m * sv[u] + qb) * sc);
      }
    }
    float sum = 0.f;
    for (int j4 = 0; j4 < 16; ++j4){
      float4 s4 = Sr[j4];
      const float* sv = (const float*)&s4;
#pragma unroll
      for (int u = 0; u < 4; ++u){
        float m = (gi || (rnd_s[j4*4+u] > ai)) ? 1.f : 0.f;
        if (inv_mask) m = 1.f - m;
        sum += __expf((m * sv[u] + qb) * sc - mx);
      }
    }
    const float inv = 1.f / sum;
    for (int j4 = 0; j4 < 16; ++j4){
      float4 s4 = Sr[j4];
      const float* sv = (const float*)&s4;
#pragma unroll
      for (int u = 0; u < 4; ++u){
        float m = (gi || (rnd_s[j4*4+u] > ai)) ? 1.f : 0.f;
        if (inv_mask) m = 1.f - m;
        W16[i*64 + j4*4 + u] = (_Float16)(m * __expf((m * sv[u] + qb) * sc - mx) * inv);
      }
    }
  }
}

__global__ __launch_bounds__(256, 1) void policy_kernel(Params p){
  extern __shared__ char smem[];
  float* S     = (float*)smem;          // 16KB f32 scores (softmax input)
  float* S1    = S  + 64*64;            // 16KB o1
  float* S2    = S1 + 64*64;            // 16KB o2
  float* rnd_s = S2 + 64*64;            // 64
  float* a_s   = rnd_s + 64;            // 64
  _Float16* X16   = (_Float16*)(a_s + 64);
  _Float16* E16   = X16   + 64*64;
  _Float16* Q16   = E16   + 64*64;
  _Float16* K16   = Q16   + 64*64;
  _Float16* VT16  = K16   + 64*64;      // V / vm stored TRANSPOSED [h][j]
  _Float16* W16   = VT16  + 64*64;
  _Float16* T16   = W16   + 64*64;      // attn_in / ctx
  _Float16* ATT16 = T16   + 64*64;
  _Float16* MSG16 = ATT16 + 64*64;
  _Float16* TMP16 = MSG16 + 64*64;
  _Float16* WB0   = TMP16 + 64*64;      // weight staging (double buffered)
  _Float16* WB1   = WB0   + 64*64;

  const int b   = blockIdx.x;
  const int tid = threadIdx.x;

  { // stage x -> X16, embed weights -> WB0, per-agent randoms
    const float* xg = p.x + (size_t)b * 64 * 64;
    const int base = tid * 16;
#pragma unroll
    for (int e = 0; e < 16; e += 4){
      const float4 v = *(const float4*)(xg + base + e);
      X16[base + e + 0] = (_Float16)v.x;
      X16[base + e + 1] = (_Float16)v.y;
      X16[base + e + 2] = (_Float16)v.z;
      X16[base + e + 3] = (_Float16)v.w;
    }
    stageW(p.embed_w, WB0);
    pf(p.ia_wq);
    if (tid < 64){
      rnd_s[tid] = p.rnd[b*64 + tid];
      a_s[tid]   = p.rnd[b*64 + (p.idx[b*64 + tid] & 63)];
    }
  }
  __syncthreads();

  // e = lrelu(x @ embed_w.T + embed_b)           | stage wq
  mmF<ACT_LRELU, ST_F16_ROW, true>(X16, WB0, p.embed_b, E16);
  stageW(p.ia_wq, WB1); pf(p.ia_wk); __syncthreads();

  // init-attn q,k,v (v stored transposed)        | stage next weight each phase
  mmF<ACT_NONE, ST_F16_ROW, true>(E16, WB1, p.ia_bq, Q16);
  stageW(p.ia_wk, WB0); pf(p.ia_wv); __syncthreads();
  mmF<ACT_NONE, ST_F16_ROW, true>(E16, WB0, p.ia_bk, K16);
  stageW(p.ia_wv, WB1); pf(p.ia_wo); __syncthreads();
  mmF<ACT_NONE, ST_F16_TR, true>(E16, WB1, p.ia_bv, VT16);
  stageW(p.ia_wo, WB0); pf(p.msg_w); __syncthreads();

  // scores (f32) -> softmax -> W16
  mmF<ACT_NONE, ST_F32_ROW, false>(Q16, K16, nullptr, S);
  stageW(p.msg_w, WB1); pf(p.st_w); __syncthreads();
  softmaxPlain(S, W16); __syncthreads();

  // attn = tanh((w @ v) @ wo.T + bo)
  mmF<ACT_NONE, ST_F16_ROW, false>(W16, VT16, nullptr, T16); __syncthreads();
  mmF<ACT_TANH, ST_F16_ROW, true>(T16, WB0, p.ia_bo, ATT16); __syncthreads();

  // messages / temp state
  mmF<ACT_NONE, ST_F16_ROW, true>(ATT16, WB1, p.msg_b, MSG16);
  stageW(p.st_w, WB0); pf(p.c1_wq); __syncthreads();
  mmF<ACT_NONE, ST_F16_ROW, true>(ATT16, WB0, p.st_b, TMP16);
  stageW(p.c1_wq, WB1); pf(p.c1_wk); __syncthreads();

  // two consensus attention passes (mg / ml masks)
  for (int pass = 0; pass < 2; ++pass){
    const float* wk = pass ? p.c2_wk : p.c1_wk; const float* bk = pass ? p.c2_bk : p.c1_bk;
    const float* wv = pass ? p.c2_wv : p.c1_wv; const float* bv = pass ? p.c2_bv : p.c1_bv;
    const float* wo = pass ? p.c2_wo : p.c1_wo; const float* bo = pass ? p.c2_bo : p.c1_bo;
    const float* bq = pass ? p.c2_bq : p.c1_bq;
    float* OUT = pass ? S2 : S1;

    // WB1 holds this pass's wq (prestaged)
    mmF<ACT_NONE, ST_F16_ROW, true>(TMP16, WB1, bq, Q16);          // qq
    stageW(wk, WB0); pf(wv); __syncthreads();
    mmF<ACT_NONE, ST_F16_ROW, false>(MSG16, WB0, nullptr, K16);    // km
    stageW(wv, WB1); pf(wo); __syncthreads();
    mmF<ACT_NONE, ST_F16_TR, false>(MSG16, WB1, nullptr, VT16);    // vm^T
    stageW(wo, WB0); __syncthreads();

    mmF<ACT_NONE, ST_F32_ROW, false>(Q16, K16, nullptr, S);        // raw qq.km^T
    if (pass == 0){ stageW(p.c2_wq, WB1); pf(p.c2_wk); }           // prestage next wq
    __syncthreads();
    softmaxMasked(S, Q16, bk, rnd_s, a_s, pass, W16); __syncthreads();

    mmF<ACT_NONE, ST_F16_ROW, true>(W16, VT16, bv, T16); __syncthreads();   // ctx
    mmF<ACT_NONE, ST_F32_ROW, true>(T16, WB0, bo, OUT); __syncthreads();    // o1/o2
  }

  // final head: lrelu(concat(o1,o2)) @ {mean_w,ls_w}.T + bias ; clip+exp for std
  {
    const size_t OutN = (size_t)gridDim.x * 64 * 8;
    const int o0 = tid * 2;            // two adjacent outputs share row i
    const int i  = o0 >> 3;
    const int a0 = o0 & 7, a1 = a0 + 1;
    float am0 = p.mean_b[a0], al0 = p.ls_b[a0];
    float am1 = p.mean_b[a1], al1 = p.ls_b[a1];
    const float4* R1 = (const float4*)(S1 + i*64);
    const float4* R2 = (const float4*)(S2 + i*64);
    for (int h4 = 0; h4 < 16; ++h4){
      float4 v1 = R1[h4], v2 = R2[h4];
      const float* f1 = (const float*)&v1;
      const float* f2 = (const float*)&v2;
#pragma unroll
      for (int u = 0; u < 4; ++u){
        const int h = h4*4 + u;
        float g1 = (f1[u] > 0.f) ? f1[u] : 0.2f*f1[u];
        float g2 = (f2[u] > 0.f) ? f2[u] : 0.2f*f2[u];
        am0 += g1 * p.mean_w[a0*128 + h] + g2 * p.mean_w[a0*128 + 64 + h];
        al0 += g1 * p.ls_w[a0*128 + h]   + g2 * p.ls_w[a0*128 + 64 + h];
        am1 += g1 * p.mean_w[a1*128 + h] + g2 * p.mean_w[a1*128 + 64 + h];
        al1 += g1 * p.ls_w[a1*128 + h]   + g2 * p.ls_w[a1*128 + 64 + h];
      }
    }
    const size_t oo = (size_t)b*512 + o0;
    p.out[oo]     = am0;
    p.out[oo + 1] = am1;
    al0 = fminf(fmaxf(al0, -20.f), 2.f);
    al1 = fminf(fmaxf(al1, -20.f), 2.f);
    p.out[OutN + oo]     = __expf(al0);
    p.out[OutN + oo + 1] = __expf(al1);
  }
}

} // namespace

extern "C" void kernel_launch(void* const* d_in, const int* in_sizes, int n_in,
                              void* d_out, int out_size, void* d_ws, size_t ws_size,
                              hipStream_t stream){
  (void)n_in; (void)out_size; (void)d_ws; (void)ws_size;
  Params p;
  p.x       = (const float*)d_in[0];
  p.rnd     = (const float*)d_in[1];
  p.idx     = (const int*)  d_in[2];
  p.embed_w = (const float*)d_in[3];  p.embed_b = (const float*)d_in[4];
  p.ia_wq = (const float*)d_in[5];  p.ia_bq = (const float*)d_in[6];
  p.ia_wk = (const float*)d_in[7];  p.ia_bk = (const float*)d_in[8];
  p.ia_wv = (const float*)d_in[9];  p.ia_bv = (const float*)d_in[10];
  p.ia_wo = (const float*)d_in[11]; p.ia_bo = (const float*)d_in[12];
  p.c1_wq = (const float*)d_in[13]; p.c1_bq = (const float*)d_in[14];
  p.c1_wk = (const float*)d_in[15]; p.c1_bk = (const float*)d_in[16];
  p.c1_wv = (const float*)d_in[17]; p.c1_bv = (const float*)d_in[18];
  p.c1_wo = (const float*)d_in[19]; p.c1_bo = (const float*)d_in[20];
  p.c2_wq = (const float*)d_in[21]; p.c2_bq = (const float*)d_in[22];
  p.c2_wk = (const float*)d_in[23]; p.c2_bk = (const float*)d_in[24];
  p.c2_wv = (const float*)d_in[25]; p.c2_bv = (const float*)d_in[26];
  p.c2_wo = (const float*)d_in[27]; p.c2_bo = (const float*)d_in[28];
  p.msg_w = (const float*)d_in[29]; p.msg_b = (const float*)d_in[30];
  p.st_w  = (const float*)d_in[31]; p.st_b  = (const float*)d_in[32];
  p.mean_w= (const float*)d_in[33]; p.mean_b= (const float*)d_in[34];
  p.ls_w  = (const float*)d_in[35]; p.ls_b  = (const float*)d_in[36];
  p.out   = (float*)d_out;

  const int B = in_sizes[0] / (64 * 64);
  const size_t shmem = 3*64*64*sizeof(float)   // S, S1, S2
                     + 2*64*sizeof(float)      // rnd_s, a_s
                     + 12*64*64*2;             // twelve f16 64x64 buffers
  hipLaunchKernelGGL(policy_kernel, dim3(B), dim3(256), shmem, stream, p);
}
